// DDPM_19207093747740
// MI455X (gfx1250) — matrix operand order repaired and checked by reading, not compiled, not run on hardware
//
#include <hip/hip_runtime.h>

typedef __attribute__((ext_vector_type(16))) _Float16 v16h;
typedef __attribute__((ext_vector_type(8)))  float    v8f;

#define N_T    100
#define IMG    128
#define NCH    64     // hidden channels
#define TILE   16     // output tile per block
#define REG    20     // h/G region = TILE + 4 (rel image offsets -2..17)
#define NXB    22     // blurred region rows/cols (rel -3..18)
#define SXB    24     // blurred region LDS stride
#define SX     24     // x region rows/cols/stride (rel -4..19)
#define NPIX   400    // REG*REG (logical pixels per channel)
#define SH     404    // sH channel stride (f16): 808B -> gcd(202,64)=2 -> conflict-free
#define NTBLK  25     // (REG/4)^2 4x4 pixel blocks (N-tiles)
#define NBLK   4096   // 64 images * 8*8 tiles

__global__ __launch_bounds__(256) void ddpm_fused(
    const float* __restrict__ x,  const int*   __restrict__ t,
    const float* __restrict__ kern, const float* __restrict__ w1,
    const float* __restrict__ b1, const float* __restrict__ w2,
    const float* __restrict__ b2, float* __restrict__ partial)
{
  __shared__ float    sX[SX * SX];        // original x tile + halo (f32)
  __shared__ _Float16 sXB[SXB * NXB];     // blurred tile (f16)
  __shared__ _Float16 sH[NCH * SH];       // hidden activations [ch][ry*20+rx], padded stride
  __shared__ _Float16 sG[9 * NPIX];       // per-tap conv2 partials
  __shared__ float    sK[9];
  __shared__ float    sB1[NCH];
  __shared__ float    sT[1];
  __shared__ float    red[256];

  const int tid  = threadIdx.x;
  const int lane = tid & 31;              // wave32
  const int wv   = tid >> 5;              // 8 waves

  const int bi  = blockIdx.x;
  const int img = bi >> 6;
  const int ty  = ((bi >> 3) & 7) * TILE;
  const int tx  = (bi & 7) * TILE;

  // ---- stage per-image kernel row, biases, t_norm ----
  if (tid < 9)   sK[tid]  = kern[t[img] * 9 + tid];
  if (tid < NCH) sB1[tid] = b1[tid];
  if (tid == 0)  sT[0]    = ((float)t[img] + 1.0f) * (1.0f / (float)N_T);

  // ---- load x tile with circular wrap (image offsets -4..19) ----
  const float* ximg = x + (size_t)img * IMG * IMG;
  for (int i = tid; i < SX * SX; i += 256) {
    int u = i / SX, ux = i % SX;
    int iy = (ty + u  - 4) & (IMG - 1);
    int ix = (tx + ux - 4) & (IMG - 1);
    sX[i] = ximg[iy * IMG + ix];
  }
  __syncthreads();

  // ---- depthwise 3x3 circular blur -> sXB (zero outside image = SAME pad for conv1) ----
  for (int i = tid; i < NXB * NXB; i += 256) {
    int s = i / NXB, sx = i % NXB;
    int iy = ty + s - 3, ix = tx + sx - 3;
    float v = 0.f;
    if ((unsigned)iy < (unsigned)IMG && (unsigned)ix < (unsigned)IMG) {
      #pragma unroll
      for (int e = 0; e < 3; ++e)
        #pragma unroll
        for (int f = 0; f < 3; ++f)
          v += sK[e * 3 + f] * sX[(s + e) * SX + (sx + f)];
    }
    sXB[s * SXB + sx] = (_Float16)v;
  }

  // ---- conv1 A operands: w1 [64x9] zero-padded to 64x32, 4 m-tiles ----
  // A layout: lane holds row M=lane%16; VGPR v elems {2v,2v+1}:
  //   v0-3: K = (lane<16 ? 0 : 8) + 2v ;  v4-7: K = (lane<16 ? 16 : 24) + 2(v-4)
  v16h a1[4];
  {
    const int  m  = lane & 15;
    const bool hi = lane >= 16;
    #pragma unroll
    for (int mt = 0; mt < 4; ++mt) {
      v16h a;
      #pragma unroll
      for (int v = 0; v < 8; ++v) {
        int k0 = (v < 4) ? ((hi ? 8 : 0) + 2 * v) : ((hi ? 24 : 16) + 2 * (v - 4));
        float e0 = (k0     < 9) ? w1[(mt * 16 + m) * 9 + k0]     : 0.f;
        float e1 = (k0 + 1 < 9) ? w1[(mt * 16 + m) * 9 + k0 + 1] : 0.f;
        a[2 * v]     = (_Float16)e0;
        a[2 * v + 1] = (_Float16)e1;
      }
      a1[mt] = a;
    }
  }
  __syncthreads();

  // ---- conv1 via WMMA: per 4x4 pixel block build B (taps x pixels), 4 m-tiles ----
  for (int nt = wv; nt < NTBLK; nt += 8) {
    const int by = (nt / 5) * 4, bx = (nt % 5) * 4;
    // B layout: lane holds pixel N=lane%16; element s = tap K (table order)
    v16h bb;
    #pragma unroll
    for (int s = 0; s < 16; ++s) {
      int kt = (s < 8) ? ((lane >> 4) * 8 + s) : (16 + (lane >> 4) * 8 + (s - 8));
      int nb = lane & 15;
      int dy = (kt < 9) ? (kt / 3 - 1) : 0;
      int dx = (kt < 9) ? (kt % 3 - 1) : 0;
      int i0 = (by + (nb >> 2) + dy + 1) * SXB + (bx + (nb & 3) + dx + 1);
      bb[s] = (kt < 9) ? sXB[i0] : (_Float16)0.0f;
    }
    const int  n    = lane & 15;
    const int  ry   = by + (n >> 2), rx = bx + (n & 3);
    const int  iy   = ty + ry - 2,   ix = tx + rx - 2;
    const bool inim = ((unsigned)iy < (unsigned)IMG) && ((unsigned)ix < (unsigned)IMG);
    #pragma unroll
    for (int mt = 0; mt < 4; ++mt) {
      v8f c = {0.f, 0.f, 0.f, 0.f, 0.f, 0.f, 0.f, 0.f};
      c = __builtin_amdgcn_wmma_f32_16x16x32_f16(false, a1[mt], false, bb,
                                                 (short)0, c, false, false);
      // D layout: lane holds pixel N=lane%16; VGPR v = channel M = v + 8*(lane/16)
      #pragma unroll
      for (int v = 0; v < 8; ++v) {
        int   ch = mt * 16 + (lane >> 4) * 8 + v;
        float hv = fmaxf(c[v] + sB1[ch] + sT[0], 0.f);
        sH[ch * SH + ry * REG + rx] = inim ? (_Float16)hv : (_Float16)0.0f;
      }
    }
  }
  __syncthreads();

  // ---- conv2 A operands: G[k][q] = sum_c w2[c][k] * h[c][q]; M=taps(9 of 16), K=64 ----
  v16h a2lo, a2hi;
  {
    const int  k  = lane & 15;
    const bool hi = lane >= 16;
    const float km = (k < 9) ? 1.f : 0.f;
    const int  kk = (k < 9) ? k : 0;
    #pragma unroll
    for (int v = 0; v < 8; ++v) {
      int c0 = (v < 4) ? ((hi ? 8 : 0) + 2 * v) : ((hi ? 24 : 16) + 2 * (v - 4));
      a2lo[2 * v]     = (_Float16)(km * w2[(c0     ) * 9 + kk]);
      a2lo[2 * v + 1] = (_Float16)(km * w2[(c0 +  1) * 9 + kk]);
      a2hi[2 * v]     = (_Float16)(km * w2[(c0 + 32) * 9 + kk]);
      a2hi[2 * v + 1] = (_Float16)(km * w2[(c0 + 33) * 9 + kk]);
    }
  }

  // ---- conv2 via WMMA: B rows = channels (lane), two K=32 chained accumulations ----
  for (int nt = wv; nt < NTBLK; nt += 8) {
    const int by = (nt / 5) * 4, bx = (nt % 5) * 4;
    v16h bl, bh;
    #pragma unroll
    for (int s = 0; s < 16; ++s) {
      int nb = lane & 15;
      int p0 = (by + (nb >> 2)) * REG + bx + (nb & 3);
      int cc = (s < 8) ? ((lane >> 4) * 8 + s) : (16 + (lane >> 4) * 8 + (s - 8));
      bl[s] = sH[cc * SH + p0];
      bh[s] = sH[(cc + 32) * SH + p0];
    }
    v8f g = {0.f, 0.f, 0.f, 0.f, 0.f, 0.f, 0.f, 0.f};
    g = __builtin_amdgcn_wmma_f32_16x16x32_f16(false, a2lo, false, bl, (short)0, g, false, false);
    g = __builtin_amdgcn_wmma_f32_16x16x32_f16(false, a2hi, false, bh, (short)0, g, false, false);
    const int n  = lane & 15;
    const int pp = (by + (n >> 2)) * REG + bx + (n & 3);
    #pragma unroll
    for (int v = 0; v < 8; ++v) {
      int k = (lane >> 4) * 8 + v;
      if (k < 9) sG[k * NPIX + pp] = (_Float16)g[v];
    }
  }
  __syncthreads();

  // ---- tap gather + bias, squared error vs original x, block reduction ----
  {
    const int py = tid >> 4, px = tid & 15;
    float acc = b2[0];
    #pragma unroll
    for (int k = 0; k < 9; ++k) {
      int qy = py + 2 + (k / 3 - 1);
      int qx = px + 2 + (k % 3 - 1);
      acc += (float)sG[k * NPIX + qy * REG + qx];
    }
    float e = sX[(py + 4) * SX + (px + 4)] - acc;
    red[tid] = e * e;
  }
  __syncthreads();
  for (int s = 128; s > 0; s >>= 1) {
    if (tid < s) red[tid] += red[tid + s];
    __syncthreads();
  }
  if (tid == 0) partial[blockIdx.x] = red[0];
}

// Deterministic fixed-order reduction of per-block partials -> scalar mean.
__global__ __launch_bounds__(256) void ddpm_reduce(const float* __restrict__ partial,
                                                   float* __restrict__ out)
{
  __shared__ float red[256];
  float s = 0.f;
  for (int i = threadIdx.x; i < NBLK; i += 256) s += partial[i];
  red[threadIdx.x] = s;
  __syncthreads();
  for (int k = 128; k > 0; k >>= 1) {
    if (threadIdx.x < k) red[threadIdx.x] += red[threadIdx.x + k];
    __syncthreads();
  }
  if (threadIdx.x == 0) out[0] = red[0] * (1.0f / (64.0f * 128.0f * 128.0f));
}

extern "C" void kernel_launch(void* const* d_in, const int* in_sizes, int n_in,
                              void* d_out, int out_size, void* d_ws, size_t ws_size,
                              hipStream_t stream) {
  const float* x    = (const float*)d_in[0];
  const int*   t    = (const int*)  d_in[1];
  const float* kern = (const float*)d_in[2];
  const float* w1   = (const float*)d_in[3];
  const float* b1   = (const float*)d_in[4];
  const float* w2   = (const float*)d_in[5];
  const float* b2   = (const float*)d_in[6];
  float* partial = (float*)d_ws;                 // 4096 floats of scratch

  ddpm_fused<<<NBLK, 256, 0, stream>>>(x, t, kern, w1, b1, w2, b2, partial);
  ddpm_reduce<<<1, 256, 0, stream>>>(partial, (float*)d_out);
}
